// SNU_Network_46772193854017
// MI455X (gfx1250) — matrix operand order repaired and checked by reading, not compiled, not run on hardware
//
#include <hip/hip_runtime.h>
#include <hip/hip_bf16.h>

typedef __attribute__((ext_vector_type(16))) __bf16 v16bf;
typedef __attribute__((ext_vector_type(8)))  __bf16 v8bf;
typedef __attribute__((ext_vector_type(8)))  float  v8f;

#define B_SZ    256
#define T_STEPS 100
#define NIN     784
#define K1PAD   800     // 784 padded to multiple of 32
#define NMID    1024
#define NOUT    10
#define TAU     0.8f
#define LDK     1024    // LDS activation row stride (elements)

union V16 { v16bf v; v8bf h[2]; };

__device__ __forceinline__ float sigmoidf(float z) {
    return 1.0f / (1.0f + __expf(-z));
}

// ---------------------------------------------------------------------------
// Weight pre-pack: W[K][N] f32 row-major  ->  Wt[Npad][Kpad] bf16 (transposed,
// zero-padded).  Kpad is the packed row stride; per-lane B fragments become
// contiguous 32-byte loads.
// ---------------------------------------------------------------------------
__global__ void pack_w(const float* __restrict__ W, __bf16* __restrict__ out,
                       int Kin, int Kpad, int Nin, int Npad) {
    int idx = blockIdx.x * blockDim.x + threadIdx.x;
    int total = Npad * Kpad;
    if (idx >= total) return;
    int n = idx / Kpad;
    int k = idx - n * Kpad;
    float v = (k < Kin && n < Nin) ? W[(size_t)k * Nin + n] : 0.0f;
    out[idx] = (__bf16)v;
}

// ---------------------------------------------------------------------------
// One SNU layer for this wave's column slice.
//   acc  = d_state (= tau*s_prev*(1-y_prev), seeds the WMMA accumulator)
//   acc += A(act_in) x B(Wt)          via v_wmma_f32_16x16x32_bf16
//   s = relu(acc); y = sigmoid(s+b); d_state = tau*s*(1-y); y -> act_out
// ---------------------------------------------------------------------------
template <int NT, bool WRITE_OUT, bool ACC_M>
__device__ __forceinline__
void snu_layer(const __bf16* __restrict__ inAct,    // LDS, row stride LDK
               int ksteps,                          // K / 32
               const __bf16* __restrict__ Wt,       // [Npad][Krow] bf16
               int Krow,
               v8f (&dstate)[NT],
               const float* bias,                   // NT per-lane bias values
               __bf16* __restrict__ outAct,         // LDS, row stride LDK
               v8f* macc,
               int n0base, int lane)
{
    const int hl   = lane >> 4;    // lane half (0/1)
    const int mrow = lane & 15;

    v8f acc[NT];
#pragma unroll
    for (int nt = 0; nt < NT; ++nt) acc[nt] = dstate[nt];

    for (int ks = 0; ks < ksteps; ++ks) {
        const int k0 = ks * 32;
        // A fragment (16x32 bf16): lanes 0-15 rows 0-15 / K 0..15,
        // lanes 16-31 same rows / K 16..31  (contiguous, matches B)
        V16 a;
        const __bf16* arow = inAct + mrow * LDK + k0 + hl * 16;
        a.h[0] = *(const v8bf*)(arow);
        a.h[1] = *(const v8bf*)(arow + 8);
#pragma unroll
        for (int nt = 0; nt < NT; ++nt) {
            // B fragment (32x16 bf16): lane n holds column n0+mrow,
            // lanes 0-15 -> K k0..k0+15, lanes 16-31 -> K k0+16..k0+31
            const __bf16* wrow =
                Wt + (size_t)(n0base + nt * 16 + mrow) * Krow + k0 + hl * 16;
            V16 b;
            b.h[0] = *(const v8bf*)(wrow);
            b.h[1] = *(const v8bf*)(wrow + 8);
            acc[nt] = __builtin_amdgcn_wmma_f32_16x16x32_bf16(
                false, a.v, false, b.v, (short)0, acc[nt], false, false);
        }
    }

#pragma unroll
    for (int nt = 0; nt < NT; ++nt) {
        const int n = n0base + nt * 16 + mrow;
#pragma unroll
        for (int r = 0; r < 8; ++r) {
            float s = fmaxf(acc[nt][r], 0.0f);
            float y = sigmoidf(s + bias[nt]);
            dstate[nt][r] = TAU * s * (1.0f - y);
            if (WRITE_OUT)
                outAct[(r + 8 * hl) * LDK + n] = (__bf16)y;   // C/D layout: row=r+8*half
            if (ACC_M)
                (*macc)[r] += y;
        }
    }
}

// ---------------------------------------------------------------------------
// Main persistent kernel: 16 workgroups (one 16-row batch tile each),
// 8 waves, each wave owns 128 output columns per mid layer.
// ---------------------------------------------------------------------------
__global__ __launch_bounds__(256)
void snu_main(const float* __restrict__ x,
              const __bf16* __restrict__ Wt1, const __bf16* __restrict__ Wt2,
              const __bf16* __restrict__ Wt3, const __bf16* __restrict__ Wt4,
              const float* __restrict__ b1, const float* __restrict__ b2,
              const float* __restrict__ b3, const float* __restrict__ b4,
              float* __restrict__ out_m)          // [B][NOUT]
{
    __shared__ __bf16 actA[16 * LDK];   // 32 KB
    __shared__ __bf16 actB[16 * LDK];   // 32 KB

    const int tid  = threadIdx.x;
    const int wave = tid >> 5;
    const int lane = tid & 31;
    const int hl   = lane >> 4;
    const int mrow = lane & 15;
    const int r0   = blockIdx.x * 16;   // batch row base
    const int n0   = wave * 128;        // column base for mid layers

    // Per-lane biases for this wave's 8 column tiles (hoisted out of T loop).
    float bia1[8], bia2[8], bia3[8];
#pragma unroll
    for (int nt = 0; nt < 8; ++nt) {
        bia1[nt] = b1[n0 + nt * 16 + mrow];
        bia2[nt] = b2[n0 + nt * 16 + mrow];
        bia3[nt] = b3[n0 + nt * 16 + mrow];
    }
    float bia4[1];
    bia4[0] = (mrow < NOUT) ? b4[mrow] : 0.0f;

    // Recurrent state (decay terms) held in accumulator-layout registers.
    v8f d1[8], d2[8], d3[8], d4[1], macc;
#pragma unroll
    for (int r = 0; r < 8; ++r) { d4[0][r] = 0.0f; macc[r] = 0.0f; }
#pragma unroll
    for (int nt = 0; nt < 8; ++nt)
#pragma unroll
        for (int r = 0; r < 8; ++r) { d1[nt][r] = 0.0f; d2[nt][r] = 0.0f; d3[nt][r] = 0.0f; }

    for (int t = 0; t < T_STEPS; ++t) {
        // Stage x_t tile into LDS (bf16), zero-padding K 784..799.
        for (int i = tid; i < 16 * K1PAD; i += 256) {
            int m = i / K1PAD;
            int k = i - m * K1PAD;
            float v = (k < NIN)
                        ? x[((size_t)(r0 + m) * T_STEPS + t) * NIN + k]
                        : 0.0f;
            actA[m * LDK + k] = (__bf16)v;
        }
        __syncthreads();

        snu_layer<8, true, false>(actA, K1PAD / 32, Wt1, K1PAD, d1, bia1, actB, nullptr, n0, lane);
        __syncthreads();
        snu_layer<8, true, false>(actB, NMID / 32, Wt2, NMID, d2, bia2, actA, nullptr, n0, lane);
        __syncthreads();
        snu_layer<8, true, false>(actA, NMID / 32, Wt3, NMID, d3, bia3, actB, nullptr, n0, lane);
        __syncthreads();

        if (wave == 0) {   // layer 4: single 16x16 tile (N padded 10->16)
            snu_layer<1, false, true>(actB, NMID / 32, Wt4, NMID, d4, bia4, actA, &macc, 0, lane);
        }
        __syncthreads();
    }

    if (wave == 0) {
#pragma unroll
        for (int r = 0; r < 8; ++r) {
            int row = r + 8 * hl;
            if (mrow < NOUT)
                out_m[(size_t)(r0 + row) * NOUT + mrow] = macc[r];
        }
    }
}

// ---------------------------------------------------------------------------
// Deterministic loss: one block, thread b handles batch row b.
// loss = mean_b( logsumexp(m_b) - m_b[y_b] )
// ---------------------------------------------------------------------------
__global__ void loss_k(const float* __restrict__ m, const long long* __restrict__ y,
                       float* __restrict__ loss_out) {
    __shared__ float red[256];
    int b = threadIdx.x;
    const float* row = m + (size_t)b * NOUT;
    float mx = row[0];
#pragma unroll
    for (int j = 1; j < NOUT; ++j) mx = fmaxf(mx, row[j]);
    float se = 0.0f;
#pragma unroll
    for (int j = 0; j < NOUT; ++j) se += __expf(row[j] - mx);
    float lse = mx + __logf(se);
    int tgt = (int)y[b];
    red[b] = lse - row[tgt];
    __syncthreads();
    for (int s = 128; s > 0; s >>= 1) {
        if (b < s) red[b] += red[b + s];
        __syncthreads();
    }
    if (b == 0) loss_out[0] = red[0] / (float)B_SZ;
}

// ---------------------------------------------------------------------------
extern "C" void kernel_launch(void* const* d_in, const int* in_sizes, int n_in,
                              void* d_out, int out_size, void* d_ws, size_t ws_size,
                              hipStream_t stream) {
    const float*     x  = (const float*)d_in[0];
    const long long* y  = (const long long*)d_in[1];
    const float*     W1 = (const float*)d_in[2];
    const float*     b1 = (const float*)d_in[3];
    const float*     W2 = (const float*)d_in[4];
    const float*     b2 = (const float*)d_in[5];
    const float*     W3 = (const float*)d_in[6];
    const float*     b3 = (const float*)d_in[7];
    const float*     W4 = (const float*)d_in[8];
    const float*     b4 = (const float*)d_in[9];
    float* out = (float*)d_out;   // [0] = loss, [1..2560] = m

    __bf16* ws  = (__bf16*)d_ws;
    __bf16* Wt1 = ws;                                   // 1024 x 800
    __bf16* Wt2 = Wt1 + (size_t)NMID * K1PAD;           // 1024 x 1024
    __bf16* Wt3 = Wt2 + (size_t)NMID * NMID;            // 1024 x 1024
    __bf16* Wt4 = Wt3 + (size_t)NMID * NMID;            // 16 x 1024

    {
        int total;
        total = NMID * K1PAD;
        pack_w<<<(total + 255) / 256, 256, 0, stream>>>(W1, Wt1, NIN, K1PAD, NMID, NMID);
        total = NMID * NMID;
        pack_w<<<(total + 255) / 256, 256, 0, stream>>>(W2, Wt2, NMID, NMID, NMID, NMID);
        pack_w<<<(total + 255) / 256, 256, 0, stream>>>(W3, Wt3, NMID, NMID, NMID, NMID);
        total = 16 * NMID;
        pack_w<<<(total + 255) / 256, 256, 0, stream>>>(W4, Wt4, NMID, NMID, NOUT, 16);
    }

    snu_main<<<B_SZ / 16, 256, 0, stream>>>(x, Wt1, Wt2, Wt3, Wt4,
                                            b1, b2, b3, b4, out + 1);
    loss_k<<<1, 256, 0, stream>>>(out + 1, y, out);
}
